// Translution_86440511800083
// MI455X (gfx1250) — matrix operand order repaired and checked by reading, not compiled, not run on hardware
//
#include <hip/hip_runtime.h>
#include <hip/hip_bf16.h>

// ---------------------------------------------------------------------------
// Translution attention for MI455X (gfx1250, wave32, WMMA).
//
// All matrix math via v_wmma_f32_16x16x32_bf16 (bf16 in, f32 accumulate).
// Batch B=16 == WMMA M, so per-token-pair relative projections are native
// 16x128x128 WMMA stacks. The 228 unique 128x128 relative matrices (22 MB
// bf16) stay L2-resident; the dots/out2 kernels recompute the relative tiles
// instead of materializing the 104 MB pre_* tensors. Small reductions
// (dots1, out1) are LDS-tiled to avoid redundant scattered L2 reads.
//
// Fragment layouts (wave32):
//  A 16x32 bf16 (ISA dense table): lane l -> M=l&15, kb=(l>=16)?8:0,
//     VGPR j: k = (j>>2)*16 + kb + 2*(j&3)  -> contiguous bf16 pair.
//  B 32x16 bf16 (same k order as A): lane l -> N=l&15,
//     kb=(l>=16)?8:0, VGPR j: k = (j>>2)*16 + kb + 2*(j&3) -> contiguous pair of BT row.
//  C 16x16 f32: VGPR r -> M = r + 8*(lane>=16), N = lane&15.
// ---------------------------------------------------------------------------

typedef __attribute__((ext_vector_type(16))) __bf16 v16bf;
typedef __attribute__((ext_vector_type(8)))  float  v8f;

#define WMMA_BF16(a, b, c) \
  __builtin_amdgcn_wmma_f32_16x16x32_bf16(false, (a), false, (b), (short)0, (c), false, false)

static __device__ __forceinline__ unsigned short f2bf(float f) {
  unsigned u = __float_as_uint(f);
  unsigned r = (u + 0x7FFFu + ((u >> 16) & 1u)) >> 16;
  return (unsigned short)r;
}

// A-fragment: A row-major MxK bf16 (stored as ushort), base = tile origin.
static __device__ __forceinline__ v16bf load_frag_A(const unsigned short* base, int ld, int lane) {
  union { v16bf v; unsigned u[8]; } f;
  const int m  = lane & 15;
  const int kb = (lane >> 4) << 3;           // 0 or 8
  const unsigned short* row = base + (size_t)m * ld;
#pragma unroll
  for (int j = 0; j < 8; ++j) {
    int k = ((j >> 2) << 4) + kb + ((j & 3) << 1);
    f.u[j] = *(const unsigned*)(row + k);    // bf16 pair (k, k+1)
  }
  return f.v;
}

// B-fragment from BT (NxK row-major bf16): column n of B == row n of BT.
static __device__ __forceinline__ v16bf load_frag_B(const unsigned short* baseT, int ld, int lane) {
  union { v16bf v; unsigned u[8]; } f;
  const int n  = lane & 15;
  const int kb = (lane >> 4) << 3;           // 0 or 8
  const unsigned short* row = baseT + (size_t)n * ld;
#pragma unroll
  for (int j = 0; j < 8; ++j) {
    f.u[j] = *(const unsigned*)(row + ((j >> 2) << 4) + kb + ((j & 3) << 1));
  }
  return f.v;
}

// ---------------------------------------------------------------------------
// Pack: f32 (bt, R, C) -> bf16 transposed (bt, C, R)
// ---------------------------------------------------------------------------
__global__ void packT_kernel(unsigned short* __restrict__ dst, const float* __restrict__ src,
                             int R, int C, long total) {
  long idx = (long)blockIdx.x * 256 + threadIdx.x;
  if (idx >= total) return;
  long rc  = (long)R * C;
  long bt  = idx / rc;
  long rem = idx - bt * rc;
  int  r   = (int)(rem / C);
  int  c   = (int)(rem - (long)r * C);
  dst[bt * rc + (long)c * R + r] = f2bf(src[idx]);
}

// ---------------------------------------------------------------------------
// LayerNorm: x (b,n,512) f32 -> xn bf16 rows ordered (n*16 + b)
// ---------------------------------------------------------------------------
__global__ void ln_kernel(const float* __restrict__ x, const float* __restrict__ g,
                          const float* __restrict__ bb, unsigned short* __restrict__ xn) {
  int tk  = blockIdx.x;            // b*65 + n
  int b   = tk / 65, n = tk - b * 65;
  int tid = threadIdx.x, w = tid >> 5, lane = tid & 31;
  const float* xr = x + (size_t)tk * 512;
  float a0 = xr[tid], a1 = xr[tid + 256];
  float s = a0 + a1, q = a0 * a0 + a1 * a1;
#pragma unroll
  for (int off = 16; off >= 1; off >>= 1) {
    s += __shfl_xor(s, off, 32);
    q += __shfl_xor(q, off, 32);
  }
  __shared__ float ss[8], qq[8];
  if (lane == 0) { ss[w] = s; qq[w] = q; }
  __syncthreads();
  if (tid == 0) {
    float S = 0.f, Q = 0.f;
    for (int i = 0; i < 8; ++i) { S += ss[i]; Q += qq[i]; }
    ss[0] = S; qq[0] = Q;
  }
  __syncthreads();
  float mean = ss[0] * (1.f / 512.f);
  float var  = qq[0] * (1.f / 512.f) - mean * mean;
  float inv  = rsqrtf(var + 1e-5f);
  size_t orow = ((size_t)n * 16 + b) * 512;
  xn[orow + tid]       = f2bf((a0 - mean) * inv * g[tid]       + bb[tid]);
  xn[orow + tid + 256] = f2bf((a1 - mean) * inv * g[tid + 256] + bb[tid + 256]);
}

// ---------------------------------------------------------------------------
// Generic WMMA GEMM: C(MxN) = A(MxK bf16 row-major) x B (via BT NxK bf16)
// 256 threads = 8 waves; wave owns one 16-wide N tile; grid=(M/16, ceil(N/128)).
// mode 0: f32 out row-major.
// mode 1: bf16 out row-major.
// mode 2: f32 out with row remap (n*16+b) -> (b*65+n) and bias add.
// mode 3: bf16 out row-major, value = 0.5*(aux[row,col] + acc)   (combine).
// ---------------------------------------------------------------------------
__global__ void gemm_wmma(const unsigned short* __restrict__ A,
                          const unsigned short* __restrict__ BT,
                          void* __restrict__ Cout, int M, int N, int K,
                          int ldc, int mode, const float* __restrict__ bias,
                          const float* __restrict__ aux) {
  int tid = threadIdx.x, w = tid >> 5, lane = tid & 31;
  int mt = blockIdx.x;
  int n0 = (blockIdx.y * 8 + w) * 16;
  if (n0 >= N) return;
  const unsigned short* Abase = A + (size_t)mt * 16 * K;
  const unsigned short* Bbase = BT + (size_t)n0 * K;
  v8f c = {};
  for (int kk = 0; kk < K; kk += 32) {
    v16bf a = load_frag_A(Abase + kk, K, lane);
    v16bf b = load_frag_B(Bbase + kk, K, lane);
    c = WMMA_BF16(a, b, c);
  }
  int col = lane & 15;
  int rb  = (lane >> 4) << 3;
  if (mode == 0) {
    float* C = (float*)Cout;
#pragma unroll
    for (int r = 0; r < 8; ++r)
      C[(size_t)(mt * 16 + rb + r) * ldc + n0 + col] = c[r];
  } else if (mode == 1) {
    unsigned short* C = (unsigned short*)Cout;
#pragma unroll
    for (int r = 0; r < 8; ++r)
      C[(size_t)(mt * 16 + rb + r) * ldc + n0 + col] = f2bf(c[r]);
  } else if (mode == 2) {
    float* C = (float*)Cout;
#pragma unroll
    for (int r = 0; r < 8; ++r) {
      int gr = mt * 16 + rb + r;
      int nt = gr >> 4, bb2 = gr & 15;
      C[((size_t)bb2 * 65 + nt) * ldc + n0 + col] = c[r] + bias[n0 + col];
    }
  } else {
    unsigned short* C = (unsigned short*)Cout;
#pragma unroll
    for (int r = 0; r < 8; ++r) {
      size_t o = (size_t)(mt * 16 + rb + r) * ldc + n0 + col;
      C[o] = f2bf(0.5f * (aux[o] + c[r]));
    }
  }
}

// ---------------------------------------------------------------------------
// dots1[b,h,n,m] = q1[b,h,n,:64] . k1[b,h,m,:64], LDS-tiled per (b,h) block.
// Writes raw dots1 into dots[bh][n][m]; dots_kernel adds dots2 and scales.
// ---------------------------------------------------------------------------
__global__ void dots1_kernel(const float* __restrict__ qkv, float* __restrict__ dots) {
  int bh = blockIdx.x;             // b*8 + h
  int b = bh >> 3, h = bh & 7;
  int tid = threadIdx.x;
  __shared__ float q1s[65 * 64];
  __shared__ float k1s[65 * 64];
  for (int i = tid; i < 65 * 64; i += 256) {
    int n = i >> 6, d = i & 63;
    size_t base = ((size_t)(n * 16 + b)) * 1536 + h * 64 + d;
    q1s[i] = qkv[base];
    k1s[i] = qkv[base + 512];
  }
  __syncthreads();
  float* drow = dots + (size_t)bh * 65 * 65;
  for (int p = tid; p < 65 * 65; p += 256) {
    int n = p / 65, m = p - n * 65;
    const float* qr = q1s + n * 64;
    const float* kr = k1s + m * 64;
    float s = 0.f;
#pragma unroll
    for (int d = 0; d < 64; ++d) s += qr[d] * kr[d];
    drow[p] = s;
  }
}

// ---------------------------------------------------------------------------
// Fused dots2 kernel: block (n, m), 8 waves = 8 heads.
// pre_q tile = xq[m] @ Wq[map[n][m]], pre_k tile = xk[n] @ Wk[map[m][n]],
// dots2 = per-head 16-column contraction of pre_q*pre_k.
// RMW: dots[b][h][n][m] = (dots1 + dots2) * 0.5 * 64^-0.5.
// ---------------------------------------------------------------------------
__global__ void dots_kernel(const unsigned short* __restrict__ xq,
                            const unsigned short* __restrict__ xk,
                            const unsigned short* __restrict__ pqT,
                            const unsigned short* __restrict__ pkT,
                            const int* __restrict__ imap,
                            float* __restrict__ dots) {
  int n = blockIdx.x, m = blockIdx.y;
  int tid = threadIdx.x, w = tid >> 5, lane = tid & 31;
  int iq = imap[n * 65 + m];
  int ik = imap[m * 65 + n];
  const unsigned short* Aq = xq + (size_t)m * 16 * 128;
  const unsigned short* Ak = xk + (size_t)n * 16 * 128;
  const unsigned short* Bq = pqT + (size_t)iq * 16384 + (size_t)(w * 16) * 128;
  const unsigned short* Bk = pkT + (size_t)ik * 16384 + (size_t)(w * 16) * 128;
  v8f cq = {}, ck = {};
#pragma unroll
  for (int kk = 0; kk < 128; kk += 32) {
    cq = WMMA_BF16(load_frag_A(Aq + kk, 128, lane), load_frag_B(Bq + kk, 128, lane), cq);
    ck = WMMA_BF16(load_frag_A(Ak + kk, 128, lane), load_frag_B(Bk + kk, 128, lane), ck);
  }
#pragma unroll
  for (int r = 0; r < 8; ++r) {
    float t = cq[r] * ck[r];
#pragma unroll
    for (int off = 8; off >= 1; off >>= 1) t += __shfl_xor(t, off, 32);
    if ((lane & 15) == 0) {                   // lanes 0 and 16 hold the sums
      int b = r + ((lane >> 4) << 3);
      size_t o = (((size_t)(b * 8 + w)) * 65 + n) * 65 + m;
      dots[o] = (dots[o] + t) * 0.0625f;      // 0.5 * 64^-0.5
    }
  }
}

// ---------------------------------------------------------------------------
// Softmax in place over last dim (65). 8320 rows, 8 rows per block.
// ---------------------------------------------------------------------------
__global__ void softmax_kernel(float* __restrict__ p) {
  int row  = blockIdx.x * 8 + (threadIdx.x >> 5);
  int lane = threadIdx.x & 31;
  float* x = p + (size_t)row * 65;
  float v0 = x[lane], v1 = x[lane + 32];
  float v2 = (lane == 0) ? x[64] : -1e30f;
  float mx = fmaxf(v0, fmaxf(v1, v2));
#pragma unroll
  for (int off = 16; off >= 1; off >>= 1) mx = fmaxf(mx, __shfl_xor(mx, off, 32));
  float e0 = __expf(v0 - mx), e1 = __expf(v1 - mx);
  float e2 = (lane == 0) ? __expf(v2 - mx) : 0.f;
  float s = e0 + e1 + e2;
#pragma unroll
  for (int off = 16; off >= 1; off >>= 1) s += __shfl_xor(s, off, 32);
  float inv = 1.f / s;
  x[lane] = e0 * inv;
  x[lane + 32] = e1 * inv;
  if (lane == 0) x[64] = e2 * inv;
}

// ---------------------------------------------------------------------------
// out1: LDS-tiled per (b,h) block. attn slab (65x65) + v1 slab (65x64) in LDS,
// each thread produces ~16 outputs of 65 MACs. Rows stored (n*16+b).
// ---------------------------------------------------------------------------
__global__ void out1_kernel(const float* __restrict__ attn, const float* __restrict__ qkv,
                            float* __restrict__ out1) {
  int bh = blockIdx.x;             // b*8 + h
  int b = bh >> 3, h = bh & 7;
  int tid = threadIdx.x;
  __shared__ float a_s[65 * 65];
  __shared__ float v_s[65 * 64];
  const float* asrc = attn + (size_t)bh * 65 * 65;
  for (int i = tid; i < 65 * 65; i += 256) a_s[i] = asrc[i];
  for (int i = tid; i < 65 * 64; i += 256) {
    int m = i >> 6, d = i & 63;
    v_s[i] = qkv[((size_t)(m * 16 + b)) * 1536 + 1024 + h * 64 + d];
  }
  __syncthreads();
  for (int o = tid; o < 65 * 64; o += 256) {
    int n = o >> 6, d = o & 63;
    const float* ar = a_s + n * 65;
    float s = 0.f;
#pragma unroll
    for (int m = 0; m < 65; ++m) s += ar[m] * v_s[m * 64 + d];
    out1[((size_t)(n * 16 + b)) * 512 + h * 64 + d] = s;
  }
}

// ---------------------------------------------------------------------------
// Fused out2: block per n, 8 waves = heads. For each m recompute the v-rel
// tile with WMMA, accumulate attn-weighted, emit bf16 rows (n*16+b).
// Prefetches next iteration's gathered Wv tile (global_prefetch_b8).
// ---------------------------------------------------------------------------
__global__ void out2_kernel(const unsigned short* __restrict__ xv,
                            const unsigned short* __restrict__ pvT,
                            const float* __restrict__ attn,
                            const int* __restrict__ imap,
                            unsigned short* __restrict__ out2p) {
  int n = blockIdx.x;
  int tid = threadIdx.x, w = tid >> 5, lane = tid & 31;
  int b0 = (lane >> 4) << 3;
  float acc[8] = {0.f, 0.f, 0.f, 0.f, 0.f, 0.f, 0.f, 0.f};
  int iv = imap[n * 65];
  for (int mm = 0; mm < 65; ++mm) {
    const unsigned short* A  = xv + (size_t)mm * 2048;                 // 16x128 tile
    const unsigned short* Bt = pvT + (size_t)iv * 16384 + (size_t)(w * 16) * 128;
    if (mm + 1 < 65) {
      iv = imap[n * 65 + mm + 1];                                      // next gather idx
      const char* pf = (const char*)(pvT + (size_t)iv * 16384 + (size_t)(w * 16) * 128)
                     + (size_t)lane * 128;                             // 32 lanes x 128B = 4KB tile
      __builtin_prefetch(pf, 0, 0);
    }
    v8f c = {};
#pragma unroll
    for (int kk = 0; kk < 128; kk += 32)
      c = WMMA_BF16(load_frag_A(A + kk, 128, lane), load_frag_B(Bt + kk, 128, lane), c);
#pragma unroll
    for (int r = 0; r < 8; ++r) {
      int b = b0 + r;
      float a = attn[(((size_t)(b * 8 + w)) * 65 + n) * 65 + mm];
      acc[r] += a * c[r];
    }
  }
  int col = lane & 15;
#pragma unroll
  for (int r = 0; r < 8; ++r)
    out2p[((size_t)(n * 16 + b0 + r)) * 128 + w * 16 + col] = f2bf(acc[r]);
}

// ---------------------------------------------------------------------------
extern "C" void kernel_launch(void* const* d_in, const int* in_sizes, int n_in,
                              void* d_out, int out_size, void* d_ws, size_t ws_size,
                              hipStream_t stream) {
  (void)in_sizes; (void)n_in; (void)out_size; (void)ws_size;
  const float* x     = (const float*)d_in[0];
  const float* ln_g  = (const float*)d_in[1];
  const float* ln_b  = (const float*)d_in[2];
  const float* w_qkv = (const float*)d_in[3];
  const float* w_q1  = (const float*)d_in[4];
  const float* w_k1  = (const float*)d_in[5];
  const float* w_v1  = (const float*)d_in[6];
  const float* p_q   = (const float*)d_in[7];
  const float* p_k   = (const float*)d_in[8];
  const float* p_v   = (const float*)d_in[9];
  const float* w_v3  = (const float*)d_in[10];
  const float* w_out = (const float*)d_in[11];
  const float* b_out = (const float*)d_in[12];
  const int*   imap  = (const int*)d_in[13];

  char* ws = (char*)d_ws;
  size_t off = 0;
  auto alloc = [&](size_t bytes) { size_t o = off; off += (bytes + 255) & ~(size_t)255; return o; };
  unsigned short* xn    = (unsigned short*)(ws + alloc((size_t)1040 * 512 * 2));
  unsigned short* wqkvT = (unsigned short*)(ws + alloc((size_t)1536 * 512 * 2));
  unsigned short* wq1T  = (unsigned short*)(ws + alloc((size_t)128 * 512 * 2));
  unsigned short* wk1T  = (unsigned short*)(ws + alloc((size_t)128 * 512 * 2));
  unsigned short* wv1T  = (unsigned short*)(ws + alloc((size_t)128 * 512 * 2));
  unsigned short* pqT   = (unsigned short*)(ws + alloc((size_t)228 * 16384 * 2));
  unsigned short* pkT   = (unsigned short*)(ws + alloc((size_t)228 * 16384 * 2));
  unsigned short* pvT   = (unsigned short*)(ws + alloc((size_t)228 * 16384 * 2));
  unsigned short* woutT = (unsigned short*)(ws + alloc((size_t)512 * 512 * 2));
  unsigned short* wv3T  = (unsigned short*)(ws + alloc((size_t)512 * 128 * 2));
  float*          qkv   = (float*)(ws + alloc((size_t)1040 * 1536 * 4));
  unsigned short* xq    = (unsigned short*)(ws + alloc((size_t)1040 * 128 * 2));
  unsigned short* xk    = (unsigned short*)(ws + alloc((size_t)1040 * 128 * 2));
  unsigned short* xv    = (unsigned short*)(ws + alloc((size_t)1040 * 128 * 2));
  float*          dots  = (float*)(ws + alloc((size_t)16 * 8 * 65 * 65 * 4));
  float*          out1  = (float*)(ws + alloc((size_t)1040 * 512 * 4));
  unsigned short* out2p = (unsigned short*)(ws + alloc((size_t)1040 * 128 * 2));
  unsigned short* ybf   = (unsigned short*)(ws + alloc((size_t)1040 * 512 * 2));

  // Weight packs (f32 -> bf16 transposed, NxK)
  packT_kernel<<<(512 * 1536 + 255) / 256, 256, 0, stream>>>(wqkvT, w_qkv, 512, 1536, 512L * 1536);
  packT_kernel<<<(512 * 128 + 255) / 256, 256, 0, stream>>>(wq1T, w_q1, 512, 128, 512L * 128);
  packT_kernel<<<(512 * 128 + 255) / 256, 256, 0, stream>>>(wk1T, w_k1, 512, 128, 512L * 128);
  packT_kernel<<<(512 * 128 + 255) / 256, 256, 0, stream>>>(wv1T, w_v1, 512, 128, 512L * 128);
  long ptot = 228L * 128 * 128;
  packT_kernel<<<(unsigned)((ptot + 255) / 256), 256, 0, stream>>>(pqT, p_q, 128, 128, ptot);
  packT_kernel<<<(unsigned)((ptot + 255) / 256), 256, 0, stream>>>(pkT, p_k, 128, 128, ptot);
  packT_kernel<<<(unsigned)((ptot + 255) / 256), 256, 0, stream>>>(pvT, p_v, 128, 128, ptot);
  packT_kernel<<<(512 * 512 + 255) / 256, 256, 0, stream>>>(woutT, w_out, 512, 512, 512L * 512);
  packT_kernel<<<(128 * 512 + 255) / 256, 256, 0, stream>>>(wv3T, w_v3, 128, 512, 128L * 512);

  // LayerNorm -> bf16 token-major rows (n*16+b)
  ln_kernel<<<1040, 256, 0, stream>>>(x, ln_g, ln_b, xn);

  // Projections
  gemm_wmma<<<dim3(65, 12), 256, 0, stream>>>(xn, wqkvT, qkv, 1040, 1536, 512, 1536, 0, nullptr, nullptr);
  gemm_wmma<<<dim3(65, 1), 256, 0, stream>>>(xn, wq1T, xq, 1040, 128, 512, 128, 1, nullptr, nullptr);
  gemm_wmma<<<dim3(65, 1), 256, 0, stream>>>(xn, wk1T, xk, 1040, 128, 512, 128, 1, nullptr, nullptr);
  gemm_wmma<<<dim3(65, 1), 256, 0, stream>>>(xn, wv1T, xv, 1040, 128, 512, 128, 1, nullptr, nullptr);

  // dots1 (LDS-tiled), then fused relative q/k projection adding dots2
  dots1_kernel<<<128, 256, 0, stream>>>(qkv, dots);
  dots_kernel<<<dim3(65, 65), 256, 0, stream>>>(xq, xk, pqT, pkT, imap, dots);

  // Softmax (in place: dots -> attn)
  softmax_kernel<<<1040, 256, 0, stream>>>(dots);

  // Attention outputs
  out1_kernel<<<128, 256, 0, stream>>>(dots, qkv, out1);
  out2_kernel<<<65, 256, 0, stream>>>(xv, pvT, dots, imap, out2p);

  // y = bf16(0.5*(out1 + out2p @ w_v3)) via WMMA (mode 3), then final
  // projection with bias and (n,b)->(b,n) scatter into d_out (mode 2).
  gemm_wmma<<<dim3(65, 4), 256, 0, stream>>>(out2p, wv3T, ybf, 1040, 512, 128, 512, 3, nullptr, out1);
  gemm_wmma<<<dim3(65, 4), 256, 0, stream>>>(ybf, woutT, (float*)d_out, 1040, 512, 512, 512, 2, b_out, nullptr);
}
